// DNC_68229850464513
// MI455X (gfx1250) — matrix operand order repaired and checked by reading, not compiled, not run on hardware
//
#include <hip/hip_runtime.h>
#include <hip/hip_bf16.h>
#include <math.h>

// ---------------- problem constants (from reference) ----------------
#define B_   32
#define T_   64
#define I_   64
#define O_   64
#define H_   512
#define N_   128
#define W_   64
#define IF_RAW 198          // 3*W + 6
#define IF_PAD 208          // padded to 13 WMMA column tiles
#define KG   640            // I + W + H   (fused gate GEMM K)
#define G4   2048           // 4*H
#define KO   576            // H + W       (output GEMM K)

typedef _Float16 h8   __attribute__((ext_vector_type(8)));
typedef _Float16 v16h __attribute__((ext_vector_type(16)));
typedef float    v8f  __attribute__((ext_vector_type(8)));

__device__ __forceinline__ float sigm(float x) { return 1.0f / (1.0f + __expf(-x)); }

// ---------------------------------------------------------------
// One 16x16 f32 tile of A(16xK) * B^T(16xK) using v_wmma_f32_16x16x32_f16.
// A source: 16 rows of lda-strided f16 (activations).
// B source: 16 rows of ldb-strided f16 (weight rows; GEMM is X @ W^T).
// A-fragment layout (ISA 7.12.2, 16-bit A 16x32): lane = {M=lane&15, khalf=lane>>4},
//   v[0..7]  = K[khalf*8 + 0..7],  v[8..15] = K[16 + khalf*8 + 0..7].
// B-fragment layout (ISA 7.12.4 B-matrix): lanes 0-15 hold K=0..7,16..23, lanes 16-31
//   hold K=8..15,24..31 for column N=lane&15 (same k order as A).
// ---------------------------------------------------------------
__device__ __forceinline__ v8f wmma_tile(const _Float16* __restrict__ Abase, int lda,
                                         const _Float16* __restrict__ Bbase, int ldb,
                                         int K, int lane) {
  const int mr = lane & 15;
  const int kh = lane >> 4;
  v8f acc = {};
  const _Float16* ap = Abase + mr * lda + kh * 8;
  const _Float16* bp = Bbase + mr * ldb + kh * 8;
  for (int k0 = 0; k0 < K; k0 += 32) {
    h8 a0 = *(const h8*)(ap + k0);        // K[kh*8 .. +8)
    h8 a1 = *(const h8*)(ap + k0 + 16);   // K[16+kh*8 .. +8)
    h8 b0 = *(const h8*)(bp + k0);        // K[kh*8 .. +8)
    h8 b1 = *(const h8*)(bp + k0 + 16);   // K[16+kh*8 .. +8)
    v16h A, Bv;
#pragma unroll
    for (int i = 0; i < 8; ++i) {
      A[i] = a0[i];  A[8 + i] = a1[i];
      Bv[i] = b0[i]; Bv[8 + i] = b1[i];
    }
    acc = __builtin_amdgcn_wmma_f32_16x16x32_f16(false, A, false, Bv,
                                                 (short)0, acc, false, false);
  }
  return acc;
}

// ---------------- one-time setup ----------------
// Fuse W_ih|W_hh into Wg16[2048][640], pad W_if to [208][512], convert W_out,
// and fold b_ih+b_hh.
__global__ void pack_weights(const float* __restrict__ W_ih, const float* __restrict__ W_hh,
                             const float* __restrict__ b_ih, const float* __restrict__ b_hh,
                             const float* __restrict__ W_if, const float* __restrict__ W_out,
                             _Float16* __restrict__ Wg16, _Float16* __restrict__ Wif16,
                             _Float16* __restrict__ Wout16, float* __restrict__ biasg) {
  const int NG = G4 * KG;
  const int NI = IF_PAD * H_;
  const int NO = O_ * KO;
  const int total = NG + NI + NO + G4;
  for (int i = blockIdx.x * blockDim.x + threadIdx.x; i < total; i += gridDim.x * blockDim.x) {
    if (i < NG) {
      int n = i / KG, k = i % KG;
      float v = (k < I_ + W_) ? W_ih[n * (I_ + W_) + k] : W_hh[n * H_ + (k - (I_ + W_))];
      Wg16[i] = (_Float16)v;
    } else if (i < NG + NI) {
      int j = i - NG; int n = j / H_, k = j % H_;
      Wif16[j] = (_Float16)((n < IF_RAW) ? W_if[n * H_ + k] : 0.0f);
    } else if (i < NG + NI + NO) {
      int j = i - NG - NI;
      Wout16[j] = (_Float16)W_out[j];
    } else {
      int n = i - NG - NI - NO;
      biasg[n] = b_ih[n] + b_hh[n];
    }
  }
}

__global__ void init_state(float* __restrict__ stateF32, int nF32,
                           _Float16* __restrict__ f16buf, int nF16) {
  for (int i = blockIdx.x * blockDim.x + threadIdx.x; i < nF32; i += gridDim.x * blockDim.x)
    stateF32[i] = 0.0f;
  for (int i = blockIdx.x * blockDim.x + threadIdx.x; i < nF16; i += gridDim.x * blockDim.x)
    f16buf[i] = (_Float16)0.0f;
}

// ---------------- per-step kernels ----------------
// inp640[b] = [ x_t(64) | rvec(64) | h(512) ]  (rvec/h written by dnc/lstm kernels)
__global__ void pack_x(const float* __restrict__ x, int t, _Float16* __restrict__ inp640) {
  int i = blockIdx.x * blockDim.x + threadIdx.x;   // 0..2047
  int b = i >> 6, ii = i & 63;
  if (b < B_) inp640[b * KG + ii] = (_Float16)x[(b * T_ + t) * I_ + ii];
}

// gates(32x2048) = inp640(32x640) @ Wg16^T + (b_ih+b_hh)
__global__ __launch_bounds__(128) void gates_step(const _Float16* __restrict__ inp640,
                                                  const _Float16* __restrict__ Wg16,
                                                  const float* __restrict__ biasg,
                                                  float* __restrict__ gates) {
  int wave = threadIdx.x >> 5, lane = threadIdx.x & 31;
  int tile = blockIdx.x * 4 + wave;          // 256 tiles = 2 x 128
  int mt = tile >> 7, nt = tile & 127;
  v8f acc = wmma_tile(inp640 + mt * 16 * KG, KG, Wg16 + nt * 16 * KG, KG, KG, lane);
  int row = mt * 16 + (lane >> 4) * 8;       // C/D layout: lanes16-31 -> M=8+r
  int col = nt * 16 + (lane & 15);
  float bias = biasg[col];
#pragma unroll
  for (int r = 0; r < 8; ++r)
    gates[(row + r) * G4 + col] = acc[r] + bias;
}

// LSTM pointwise; writes h as f16 directly into next-step WMMA A operands.
__global__ __launch_bounds__(512) void lstm_step(const float* __restrict__ gates,
                                                 float* __restrict__ c,
                                                 _Float16* __restrict__ inp640,
                                                 _Float16* __restrict__ hr576) {
  int b = blockIdx.x, j = threadIdx.x;
  const float* g = gates + b * G4;
  float ig = sigm(g[j]);
  float fg = sigm(g[H_ + j]);
  float gg = tanhf(g[2 * H_ + j]);
  float og = sigm(g[3 * H_ + j]);
  float cn = fg * c[b * H_ + j] + ig * gg;
  c[b * H_ + j] = cn;
  _Float16 hf = (_Float16)(og * tanhf(cn));
  inp640[b * KG + I_ + W_ + j] = hf;   // gate GEMM input for step t+1
  hr576[b * KO + j] = hf;              // interface / output GEMM input for step t
}

// itf(32x208) = h(32x512) @ Wif16^T + b_if   (cols >= 198 are zero padding)
__global__ __launch_bounds__(128) void itf_step(const _Float16* __restrict__ hr576,
                                                const _Float16* __restrict__ Wif16,
                                                const float* __restrict__ b_if,
                                                float* __restrict__ itf) {
  int wave = threadIdx.x >> 5, lane = threadIdx.x & 31;
  int tile = blockIdx.x * 4 + wave;
  if (tile >= 2 * (IF_PAD / 16)) return;      // 26 tiles
  int mt = tile / (IF_PAD / 16), nt = tile % (IF_PAD / 16);
  v8f acc = wmma_tile(hr576 + mt * 16 * KO, KO, Wif16 + nt * 16 * H_, H_, H_, lane);
  int row = mt * 16 + (lane >> 4) * 8;
  int col = nt * 16 + (lane & 15);
  float bias = (col < IF_RAW) ? b_if[col] : 0.0f;
#pragma unroll
  for (int r = 0; r < 8; ++r)
    itf[(row + r) * IF_PAD + col] = acc[r] + bias;
}

// DNC memory update: one workgroup per batch sample, one thread per slot n.
// Stable argsort of usage replaced by the equivalent predecessor-product:
//   alloc[p] = (1-u[p]) * prod_{q stably-before p} u[q].
__global__ __launch_bounds__(128) void dnc_step(const float* __restrict__ itf,
    float* __restrict__ mem, float* __restrict__ usage, float* __restrict__ prec,
    float* __restrict__ link, float* __restrict__ rw,
    _Float16* __restrict__ inp640, _Float16* __restrict__ hr576) {
  __shared__ float su[N_], sww[N_], sprec[N_], srw[N_], slog[N_], sexp[N_], swts[N_], sbw[N_];
  __shared__ float serase[W_], srkey[W_], swvec[W_];
  __shared__ float sc[6];   // wg, ag, rstr, rmode0..2
  int b = blockIdx.x, n = threadIdx.x;
  const float* it = itf + b * IF_PAD;
  if (n < W_) {
    float raw = it[n];
    srkey[n]  = raw;           // raw read key
    serase[n] = sigm(raw);     // erase vector = sigmoid(same slots)
    swvec[n]  = it[W_ + n];
  }
  if (n == 0) {
    sc[0] = sigm(it[2 * W_]);          // write gate
    sc[1] = sigm(it[2 * W_ + 1]);      // alloc gate
    float xs = it[3 * W_ + 2];
    sc[2] = (xs > 20.0f) ? xs : log1pf(__expf(xs));   // read strength (softplus)
    float m0 = it[3 * W_ + 3], m1 = it[3 * W_ + 4], m2 = it[3 * W_ + 5];
    float mx = fmaxf(m0, fmaxf(m1, m2));
    float e0 = __expf(m0 - mx), e1 = __expf(m1 - mx), e2 = __expf(m2 - mx);
    float s = e0 + e1 + e2;
    sc[3] = e0 / s; sc[4] = e1 / s; sc[5] = e2 / s;   // read mode
  }
  su[n]    = usage[b * N_ + n];
  sprec[n] = prec[b * N_ + n];
  srw[n]   = rw[b * N_ + n];
  sbw[n]   = 0.0f;
  __syncthreads();

  // allocation weighting without an explicit sort
  float un = su[n];
  float p = 1.0f;
  for (int q = 0; q < N_; ++q) {
    float uq = su[q];
    bool before = (uq < un) || ((uq == un) && (q < n));
    p *= before ? uq : 1.0f;
  }
  float wwn = sc[0] * sc[1] * (1.0f - un) * p;
  sww[n] = wwn;
  __syncthreads();
  float wwsum = 0.0f;
  for (int q = 0; q < N_; ++q) wwsum += sww[q];

  // memory row update + content-similarity logit
  float dotn = 0.0f, mn2 = 0.0f, kn2 = 0.0f;
  float* mr = mem + (size_t)(b * N_ + n) * W_;
#pragma unroll 8
  for (int w = 0; w < W_; ++w) {
    float mv = mr[w];
    mv = mv * (1.0f - wwn * serase[w]) + wwn * swvec[w];
    mr[w] = mv;
    dotn += srkey[w] * mv;
    mn2  += mv * mv;
    kn2  += srkey[w] * srkey[w];
  }
  slog[n] = dotn / ((sqrtf(kn2) + 1e-8f) * (sqrtf(mn2) + 1e-8f)) * sc[2];
  usage[b * N_ + n] = un + (1.0f - un) * wwn;

  // temporal link matrix row + forward weight; backward via LDS float atomics
  float fwn = 0.0f;
  float pn  = sprec[n];
  float rwn = srw[n];
  float* lr = link + (size_t)(b * N_ + n) * N_;
  for (int j = 0; j < N_; ++j) {
    float l = lr[j];
    l = (1.0f - wwn - sww[j]) * l + pn * sww[j];
    l = (j == n) ? 0.0f : l;
    lr[j] = l;
    fwn += l * srw[j];                 // fw[n] = sum_m L[n][m]*rw[m]
    atomicAdd(&sbw[j], l * rwn);       // bw[m] = sum_n L[n][m]*rw[n]
  }
  prec[b * N_ + n] = (1.0f - wwsum) * pn + wwn;
  __syncthreads();

  // content softmax over slots
  float mx = -3.4e38f;
  for (int q = 0; q < N_; ++q) mx = fmaxf(mx, slog[q]);
  float en = __expf(slog[n] - mx);
  sexp[n] = en;
  __syncthreads();
  float esum = 0.0f;
  for (int q = 0; q < N_; ++q) esum += sexp[q];
  float cwn = en / esum;

  float wtn = sc[3] * sbw[n] + sc[4] * fwn + sc[5] * cwn + 1e-8f;
  swts[n] = wtn;
  __syncthreads();
  float wsum = 0.0f;
  for (int q = 0; q < N_; ++q) wsum += swts[q];
  wtn = wtn / wsum;
  rw[b * N_ + n] = wtn;
  slog[n] = wtn;                       // slog reads are all done; reuse as wts
  __syncthreads();

  // read vector, emitted straight as f16 GEMM operands
  if (n < W_) {
    float rv = 0.0f;
    for (int q = 0; q < N_; ++q)
      rv += slog[q] * mem[(size_t)(b * N_ + q) * W_ + n];
    _Float16 rf = (_Float16)rv;
    inp640[b * KG + I_ + n]  = rf;     // gate GEMM input for step t+1
    hr576[b * KO + H_ + n]   = rf;     // output GEMM input for this step
  }
}

// out[b][t][:] = [h|rvec](32x576) @ Wout16^T + b_out
__global__ __launch_bounds__(128) void out_step(const _Float16* __restrict__ hr576,
                                                const _Float16* __restrict__ Wout16,
                                                const float* __restrict__ b_out,
                                                float* __restrict__ out, int t) {
  int wave = threadIdx.x >> 5, lane = threadIdx.x & 31;
  int tile = blockIdx.x * 4 + wave;          // 8 tiles = 2 x 4
  int mt = tile >> 2, nt = tile & 3;
  v8f acc = wmma_tile(hr576 + mt * 16 * KO, KO, Wout16 + nt * 16 * KO, KO, KO, lane);
  int row = mt * 16 + (lane >> 4) * 8;
  int col = nt * 16 + (lane & 15);
  float bias = b_out[col];
#pragma unroll
  for (int r = 0; r < 8; ++r)
    out[((row + r) * T_ + t) * O_ + col] = acc[r] + bias;
}

// ---------------- host orchestration ----------------
extern "C" void kernel_launch(void* const* d_in, const int* in_sizes, int n_in,
                              void* d_out, int out_size, void* d_ws, size_t ws_size,
                              hipStream_t stream) {
  (void)in_sizes; (void)n_in; (void)out_size; (void)ws_size;
  const float* x     = (const float*)d_in[0];
  const float* W_ih  = (const float*)d_in[1];
  const float* W_hh  = (const float*)d_in[2];
  const float* b_ih  = (const float*)d_in[3];
  const float* b_hh  = (const float*)d_in[4];
  const float* W_if  = (const float*)d_in[5];
  const float* b_if  = (const float*)d_in[6];
  const float* W_out = (const float*)d_in[7];
  const float* b_out = (const float*)d_in[8];
  float* out = (float*)d_out;

  char* ws = (char*)d_ws;
  size_t off = 0;
  auto alloc = [&](size_t bytes) -> void* {
    void* p = (void*)(ws + off);
    off = (off + bytes + 255) & ~(size_t)255;
    return p;
  };
  _Float16* Wg16   = (_Float16*)alloc((size_t)G4 * KG * 2);      // 2.6 MB (L2-resident)
  _Float16* Wif16  = (_Float16*)alloc((size_t)IF_PAD * H_ * 2);
  _Float16* Wout16 = (_Float16*)alloc((size_t)O_ * KO * 2);
  float*    biasg  = (float*)alloc((size_t)G4 * 4);

  const size_t nC = (size_t)B_ * H_, nMem = (size_t)B_ * N_ * W_;
  const size_t nU = (size_t)B_ * N_, nP = (size_t)B_ * N_;
  const size_t nL = (size_t)B_ * N_ * N_, nR = (size_t)B_ * N_;
  const size_t nState = nC + nMem + nU + nP + nL + nR;
  float* stateF32 = (float*)alloc(nState * 4);
  float* c_st   = stateF32;
  float* mem_st = c_st + nC;
  float* u_st   = mem_st + nMem;
  float* p_st   = u_st + nU;
  float* l_st   = p_st + nP;
  float* rw_st  = l_st + nL;

  const size_t nF16 = (size_t)B_ * KG + (size_t)B_ * KO;
  _Float16* f16buf = (_Float16*)alloc(nF16 * 2);
  _Float16* inp640 = f16buf;                       // [32][640]
  _Float16* hr576  = f16buf + (size_t)B_ * KG;     // [32][576]

  float* gates = (float*)alloc((size_t)B_ * G4 * 4);
  float* itf   = (float*)alloc((size_t)B_ * IF_PAD * 4);

  pack_weights<<<512, 256, 0, stream>>>(W_ih, W_hh, b_ih, b_hh, W_if, W_out,
                                        Wg16, Wif16, Wout16, biasg);
  init_state<<<512, 256, 0, stream>>>(stateF32, (int)nState, f16buf, (int)nF16);

  for (int t = 0; t < T_; ++t) {
    pack_x<<<8, 256, 0, stream>>>(x, t, inp640);
    gates_step<<<64, 128, 0, stream>>>(inp640, Wg16, biasg, gates);
    lstm_step<<<B_, H_, 0, stream>>>(gates, c_st, inp640, hr576);
    itf_step<<<7, 128, 0, stream>>>(hr576, Wif16, b_if, itf);
    dnc_step<<<B_, N_, 0, stream>>>(itf, mem_st, u_st, p_st, l_st, rw_st, inp640, hr576);
    out_step<<<2, 128, 0, stream>>>(hr576, Wout16, b_out, out, t);
  }
}
